// ScaledDotProductAttentionwithEdge2_7834020348636
// MI455X (gfx1250) — matrix operand order repaired and checked by reading, not compiled, not run on hardware
//
#include <hip/hip_runtime.h>
#include <hip/hip_bf16.h>

typedef __attribute__((ext_vector_type(16))) _Float16 v16h;
typedef __attribute__((ext_vector_type(8)))  float    v8f;
typedef __attribute__((ext_vector_type(4)))  float    v4f;
typedef __attribute__((ext_vector_type(4)))  int      v4i;

#define NEG_INF_F (-1.0e9f)

// One wave (32 lanes) processes one node (b,h,n):
//   scores = Kmat(32x32) * q(32) / sqrt(32)   -> 2x v_wmma_f32_16x16x32_f16
//   softmax over 32 (masked)                  -> per-lane + 1x shfl_xor(16)
//   out    = attn(32) * V(32x32)              -> 2x v_wmma_f32_16x16x32_f16
__global__ __launch_bounds__(256, 1) void edge_attn_wmma(
    const float* __restrict__ Q,   // [nodes, 32]
    const float* __restrict__ K,   // [nodes, 32, 32]
    const float* __restrict__ V,   // [nodes, 32, 32]
    const int*   __restrict__ M,   // [nodes, 32]
    float* __restrict__ O,         // [nodes, 32]
    int nodes)
{
    const int lane = threadIdx.x & 31;
    const int wid  = threadIdx.x >> 5;
    const int wpb  = blockDim.x >> 5;
    const int half = lane >> 4;        // 0: lanes 0-15, 1: lanes 16-31
    const int l16  = lane & 15;
    const float scale = 0.17677669529663687f;   // 1/sqrt(32)

    long node = (long)blockIdx.x * wpb + wid;
    const long stride = (long)gridDim.x * wpb;

    for (; node < (long)nodes; node += stride) {
        const float* qn = Q + node * 32;
        const float* kn = K + node * 1024;
        const float* vn = V + node * 1024;
        const int*   mn = M + node * 32;

        // Prefetch next iteration's key/value tiles (global_prefetch_b8).
        {
            long nn = node + stride;
            if (nn < (long)nodes) {
                __builtin_prefetch((const char*)(K + nn * 1024) + lane * 256, 0, 0);
                __builtin_prefetch((const char*)(V + nn * 1024) + lane * 256, 0, 0);
            }
        }

        // ---- B operand: q replicated across all 16 columns.
        // 16-bit B 32x16 layout: lanes 0-15 hold d = {0..7, 16..23}, lanes 16-31 hold d = {8..15, 24..31}.
        v16h bq;
        {
            const v4f* q4 = (const v4f*)(qn + 8 * half);
            #pragma unroll
            for (int c = 0; c < 4; ++c) {
                v4f t = q4[c + (c & 2)];
                #pragma unroll
                for (int j = 0; j < 4; ++j) bq[4 * c + j] = (_Float16)t[j];
            }
        }

        // ---- A operand: key rows. 16-bit A 16x32 layout:
        // lanes 0-15: row m=l16, d = {0..7, 16..23}; lanes 16-31: row m=l16, d = {8..15, 24..31}.
        v16h a0, a1;
        {
            const int dbase = half * 8;
            const float* r0 = kn + (size_t)l16 * 32 + dbase;         // rows 0..15
            const float* r1 = kn + (size_t)(l16 + 16) * 32 + dbase;  // rows 16..31
            v4f k00 = *(const v4f*)(r0);
            v4f k01 = *(const v4f*)(r0 + 4);
            v4f k02 = *(const v4f*)(r0 + 16);
            v4f k03 = *(const v4f*)(r0 + 20);
            v4f k10 = *(const v4f*)(r1);
            v4f k11 = *(const v4f*)(r1 + 4);
            v4f k12 = *(const v4f*)(r1 + 16);
            v4f k13 = *(const v4f*)(r1 + 20);
            #pragma unroll
            for (int j = 0; j < 4; ++j) {
                a0[j]      = (_Float16)k00[j];
                a0[4 + j]  = (_Float16)k01[j];
                a0[8 + j]  = (_Float16)k02[j];
                a0[12 + j] = (_Float16)k03[j];
                a1[j]      = (_Float16)k10[j];
                a1[4 + j]  = (_Float16)k11[j];
                a1[8 + j]  = (_Float16)k12[j];
                a1[12 + j] = (_Float16)k13[j];
            }
        }

        // ---- scores: two 16-row blocks. C layout: lane holds k = v + 8*half (+16 for block 1).
        v8f zacc = {};
        v8f c0 = __builtin_amdgcn_wmma_f32_16x16x32_f16(false, a0, false, bq,
                                                        (short)0, zacc, false, false);
        v8f c1 = __builtin_amdgcn_wmma_f32_16x16x32_f16(false, a1, false, bq,
                                                        (short)0, zacc, false, false);

        // ---- mask + scale (this lane's 16 k-indices)
        v4i m0a = *(const v4i*)(mn + 8 * half);
        v4i m0b = *(const v4i*)(mn + 8 * half + 4);
        v4i m1a = *(const v4i*)(mn + 16 + 8 * half);
        v4i m1b = *(const v4i*)(mn + 16 + 8 * half + 4);

        float sc[16];
        #pragma unroll
        for (int j = 0; j < 4; ++j) {
            sc[j]      = (m0a[j] != 0) ? c0[j]     * scale : NEG_INF_F;
            sc[4 + j]  = (m0b[j] != 0) ? c0[4 + j] * scale : NEG_INF_F;
            sc[8 + j]  = (m1a[j] != 0) ? c1[j]     * scale : NEG_INF_F;
            sc[12 + j] = (m1b[j] != 0) ? c1[4 + j] * scale : NEG_INF_F;
        }

        // ---- softmax over 32: 16 per-lane values + cross-half combine
        float mx = sc[0];
        #pragma unroll
        for (int i = 1; i < 16; ++i) mx = fmaxf(mx, sc[i]);
        mx = fmaxf(mx, __shfl_xor(mx, 16, 32));

        float ev[16];
        float s = 0.0f;
        #pragma unroll
        for (int i = 0; i < 16; ++i) { ev[i] = __expf(sc[i] - mx); s += ev[i]; }
        s += __shfl_xor(s, 16, 32);
        const float invs = 1.0f / s;

        // ---- A operand for output GEMM: score-C layout == 16-bit A layout, direct repack.
        v16h aw;
        #pragma unroll
        for (int e = 0; e < 16; ++e) aw[e] = (_Float16)ev[e];

        // ---- B operand: V columns. Block cb covers output cols d = 16*cb + l16.
        v16h bv0, bv1;
        {
            const float* vb = vn + l16;
            #pragma unroll
            for (int e = 0; e < 16; ++e) {
                const float* rowp = vb + (size_t)(8 * half + e + (e & 8)) * 32;
                bv0[e] = (_Float16)rowp[0];    // cols 0..15
                bv1[e] = (_Float16)rowp[16];   // cols 16..31
            }
        }

        v8f o0 = __builtin_amdgcn_wmma_f32_16x16x32_f16(false, aw, false, bv0,
                                                        (short)0, zacc, false, false);
        v8f o1 = __builtin_amdgcn_wmma_f32_16x16x32_f16(false, aw, false, bv1,
                                                        (short)0, zacc, false, false);

        // All A rows are identical, so C VGPR0 holds the result row.
        // lanes 0-15 take col l16 from block 0; lanes 16-31 take col 16+l16 from block 1.
        float res = (half ? o1[0] : o0[0]) * invs;
        O[node * 32 + lane] = res;
    }
}

extern "C" void kernel_launch(void* const* d_in, const int* in_sizes, int n_in,
                              void* d_out, int out_size, void* d_ws, size_t ws_size,
                              hipStream_t stream) {
    const float* Q = (const float*)d_in[0];
    const float* K = (const float*)d_in[1];
    const float* V = (const float*)d_in[2];
    const int*   M = (const int*)d_in[3];
    float* O = (float*)d_out;

    const int nodes = in_sizes[0] / 32;   // B*H*N = 320000

    const int threads = 256;              // 8 waves / block
    const int blocks = 4096;              // grid-stride; ~10 nodes per wave
    edge_attn_wmma<<<blocks, threads, 0, stream>>>(Q, K, V, M, O, nodes);
}
